// LagPhysResGRUModel_60387240181794
// MI455X (gfx1250) — matrix operand order repaired and checked by reading, compiled, not run on hardware
//
#include <hip/hip_runtime.h>
#include <cmath>

// ---------------------------------------------------------------------------
// LagPhysResGRU fused sequence kernel for MI455X (gfx1250, wave32, WMMA)
//
// Persistent per-batch-tile workgroups (32 rows each, 128 WGs); all 48 steps
// run inside the kernel with carry state in LDS.  GEMMs use
// V_WMMA_F32_16X16X32_F16 with f32 accumulation.  Weights are packed once per
// launch into B-fragment layout in d_ws (two coalesced 128-bit loads per lane
// per tile).  Wave id is readfirstlane'd so all branches around WMMA are
// scalar (EXEC stays all-ones, as the ISA requires for WMMA).
// ---------------------------------------------------------------------------

typedef __attribute__((ext_vector_type(16))) _Float16 v16h;
typedef __attribute__((ext_vector_type(8)))  float    v8f;

#define B_TOT   4096
#define T_STEPS 48
#define SDIM    12
#define CDIM    4
#define HDIM    128
#define MLPH    1024
#define BT      32      // batch rows per workgroup
#define NW      8       // waves per block (256 threads, wave32)

#define DT_C    0.01f
#define GRAV_C  9.81f
#define KT_C    0.1f
#define KC_C    0.01f
#define ARM_C   0.046f
#define EPS_C   1e-12f

// packed-weight offsets in d_ws (units: _Float16 elements)
//   block = kt*NTILES + nt ; each block = 32 lanes * 16 halves = 512
#define OFF_WIH 0          // 1  ktile  x 24 ntiles  (K=16 padded to 32, N=384)
#define OFF_WHH 12288      // 4  ktiles x 24 ntiles  (K=128, N=384)
#define OFF_W1  61440      // 5  ktiles x 64 ntiles  (K=152 padded 160, N=1024)
#define OFF_W2  225280     // 32 ktiles x 64 ntiles  (K=1024, N=1024)
#define OFF_WO  1273856    // 32 ktiles x 1  ntile   (K=1024, N=12 padded 16)

__device__ __forceinline__ float sigmoidf_(float x) { return 1.0f / (1.0f + expf(-x)); }

__device__ __forceinline__ v8f wmma_f16(v16h a, v16h b, v8f c) {
    return __builtin_amdgcn_wmma_f32_16x16x32_f16(false, a, false, b,
                                                  (short)0, c, false, false);
}

// A-fragment (16x32 f16) packed address for element (row, k) of a [32 x 32*KT]
// activation matrix.  Matches ISA 16-bit A layout:
//   k = ((p>>3)<<4) | ((lane>>4)<<3) | (p&7),  row = lane&15  (+16*mtile)
__device__ __forceinline__ int apack_off(int row, int k, int KT) {
    int mt   = row >> 4;
    int kt   = k >> 5;
    int kin  = k & 31;
    int lane = (row & 15) | (((kin >> 3) & 1) << 4);
    int p    = (((kin >> 4) & 1) << 3) | (kin & 7);
    return (((mt * KT) + kt) * 32 + lane) * 16 + p;
}

// -------------------------- weight pack pre-kernel -------------------------
// B-fragment (32x16 f16): k = kt*32 + ((p>>3)<<4) + ((lane>>4)<<3) + (p&7) ; n = nt*16 + (lane&15)
__global__ void lagphys_pack_w(const float* __restrict__ src, _Float16* __restrict__ dst,
                               int K, int N, int NTILES, int total) {
    for (int i = blockIdx.x * blockDim.x + threadIdx.x; i < total;
         i += gridDim.x * blockDim.x) {
        int p    = i & 15;
        int lane = (i >> 4) & 31;
        int blk  = i >> 9;
        int kt   = blk / NTILES;
        int nt   = blk - kt * NTILES;
        int k    = kt * 32 + ((p >> 3) << 4) + ((lane >> 4) << 3) + (p & 7);
        int n    = nt * 16 + (lane & 15);
        float v  = (k < K && n < N) ? src[(size_t)k * N + n] : 0.0f;
        dst[i] = (_Float16)v;
    }
}

// ------------------------------ main kernel --------------------------------
__global__ __launch_bounds__(256)
void lagphys_fused_seq(const float* __restrict__ x0,
                       const float* __restrict__ u_seq,
                       const float* __restrict__ logit_alpha,
                       const float* __restrict__ hinit_W,
                       const float* __restrict__ hinit_b,
                       const float* __restrict__ gru_bih,
                       const float* __restrict__ gru_bhh,
                       const float* __restrict__ alpha_W,
                       const float* __restrict__ alpha_b,
                       const float* __restrict__ mlp_b1,
                       const float* __restrict__ mlp_b2,
                       const float* __restrict__ out_b,
                       const _Float16* __restrict__ wpack,
                       float* __restrict__ out) {
    const int tid  = threadIdx.x;
    // scalar wave id -> scalar branches, EXEC untouched around WMMA
    const int w    = __builtin_amdgcn_readfirstlane(tid >> 5);
    const int lane = tid & 31;
    const int b0   = blockIdx.x * BT;
    const size_t lb = (size_t)lane << 4;   // lane*16 halves

    const _Float16* WihP = wpack + OFF_WIH;
    const _Float16* WhhP = wpack + OFF_WHH;
    const _Float16* W1P  = wpack + OFF_W1;
    const _Float16* W2P  = wpack + OFF_W2;
    const _Float16* WoP  = wpack + OFF_WO;

    // ---- LDS ----  total ~137.5 KB
    __shared__ __align__(32) unsigned char s_big[96 * 1024]; // gi|gh f32  /  hidA f16
    __shared__ __align__(32) _Float16 s_hA[2 * 4 * 32 * 16];    // h   (32x128) A-packed
    __shared__ __align__(32) _Float16 s_xuA[2 * 1 * 32 * 16];   // xu  (32x32)  A-packed
    __shared__ __align__(32) _Float16 s_featA[2 * 5 * 32 * 16]; // feat(32x160) A-packed
    __shared__ float s_hf[BT * HDIM];
    __shared__ float s_xn[BT * SDIM];
    __shared__ float s_xp[BT * SDIM];
    __shared__ float s_xnext[BT * SDIM];
    __shared__ float s_uraw[BT * CDIM];
    __shared__ float s_ueff[BT * CDIM];

    float*    s_gi   = (float*)s_big;                    // 32x384
    float*    s_gh   = (float*)(s_big + 48 * 1024);      // 32x384
    _Float16* s_hidA = (_Float16*)s_big;                 // 32x1024 A-packed (KT=32)

    // ---------------- init ----------------
    for (int e = tid; e < BT * SDIM; e += 256) {
        int r = e / SDIM, c = e - r * SDIM;
        s_xn[e] = x0[(b0 + r) * SDIM + c];
    }
    for (int e = tid; e < BT * CDIM; e += 256) {
        int r = e >> 2, c = e & 3;
        s_ueff[e] = u_seq[((b0 + r) * T_STEPS + 0) * CDIM + c];
    }
    for (int e = tid; e < BT * 8; e += 256) {            // zero feat pad cols 152..159
        int r = e >> 3, c = 152 + (e & 7);
        s_featA[apack_off(r, c, 5)] = (_Float16)0.0f;
    }
    __syncthreads();
    for (int e = tid; e < BT * HDIM; e += 256) {         // h0 = tanh(x0 @ hinit_W + b)
        int r = e >> 7, c = e & 127;
        float acc = hinit_b[c];
        for (int k = 0; k < SDIM; ++k) acc += s_xn[r * SDIM + k] * hinit_W[k * HDIM + c];
        float h = tanhf(acc);
        s_hf[e] = h;
        s_hA[apack_off(r, c, 4)] = (_Float16)h;
    }
    __syncthreads();

    // ---------------- time loop ----------------
    for (int t = 0; t < T_STEPS; ++t) {
        // pass A: load u_raw, build xu = [x_norm | u_raw | 0-pad] (K padded to 32)
        for (int e = tid; e < BT * 32; e += 256) {
            int r = e >> 5, k = e & 31;
            float v;
            if (k < SDIM) v = s_xn[r * SDIM + k];
            else if (k < 16) {
                v = u_seq[((b0 + r) * T_STEPS + t) * CDIM + (k - SDIM)];
                s_uraw[r * CDIM + (k - SDIM)] = v;
            } else v = 0.0f;
            s_xuA[apack_off(r, k, 1)] = (_Float16)v;
        }
        __syncthreads();

        // GEMM1a: gi = xu@Wih  (48 tiles, 1 K-step each; uniform loop, no divergence)
        for (int job = w; job < 48; job += NW) {
            int mt = job / 24, nt = job - mt * 24;
            v16h a = *(const v16h*)(s_xuA + (size_t)mt * 512 + lb);
            v16h b = *(const v16h*)(WihP + ((size_t)nt << 9) + lb);
            v8f acc = {};
            acc = wmma_f16(a, b, acc);
            int rbase = mt * 16 + ((lane >> 4) << 3);
            int col   = nt * 16 + (lane & 15);
            #pragma unroll
            for (int i = 0; i < 8; ++i) s_gi[(rbase + i) * 384 + col] = acc[i];
        }
        // GEMM1b: gh = h@Whh  (48 tiles, 4 K-steps each)
        for (int job = w; job < 48; job += NW) {
            int mt = job / 24, nt = job - mt * 24;
            const _Float16* ap = s_hA + (size_t)mt * (4 * 512) + lb;
            const _Float16* bp = WhhP + ((size_t)nt << 9) + lb;
            v8f acc = {};
            #pragma unroll
            for (int kt = 0; kt < 4; ++kt) {
                v16h a = *(const v16h*)ap;
                v16h b = *(const v16h*)bp;
                acc = wmma_f16(a, b, acc);
                ap += 512;
                bp += 24 * 512;
            }
            int rbase = mt * 16 + ((lane >> 4) << 3);
            int col   = nt * 16 + (lane & 15);
            #pragma unroll
            for (int i = 0; i < 8; ++i) s_gh[(rbase + i) * 384 + col] = acc[i];
        }
        __syncthreads();

        // GRU nonlinearity -> h_new ; write f16 copies for next GEMMs
        for (int e = tid; e < BT * HDIM; e += 256) {
            int r = e >> 7, c = e & 127;
            float ir  = s_gi[r * 384 + c]       + gru_bih[c];
            float iz  = s_gi[r * 384 + c + 128] + gru_bih[c + 128];
            float inn = s_gi[r * 384 + c + 256] + gru_bih[c + 256];
            float hr  = s_gh[r * 384 + c]       + gru_bhh[c];
            float hz  = s_gh[r * 384 + c + 128] + gru_bhh[c + 128];
            float hn  = s_gh[r * 384 + c + 256] + gru_bhh[c + 256];
            float rg = sigmoidf_(ir + hr);
            float zg = sigmoidf_(iz + hz);
            float ng = tanhf(inn + rg * hn);
            float hnew = (1.0f - zg) * ng + zg * s_hf[e];
            s_hf[e] = hnew;
            _Float16 hh = (_Float16)hnew;
            s_hA[apack_off(r, c, 4)] = hh;
            s_featA[apack_off(r, 24 + c, 5)] = hh;
        }
        __syncthreads();

        // per-row: alpha gate, u_eff blend, physics step, feat head
        if (tid < BT) {
            int r = tid;
            float dot = alpha_b[0];
            for (int k = 0; k < HDIM; ++k) dot += s_hf[r * HDIM + k] * alpha_W[k];
            float ue[4], ur[4];
            #pragma unroll
            for (int j = 0; j < 4; ++j) {
                float al = sigmoidf_(logit_alpha[j] + dot);
                ur[j] = s_uraw[r * CDIM + j];
                ue[j] = al * s_ueff[r * CDIM + j] + (1.0f - al) * ur[j];
                s_ueff[r * CDIM + j] = ue[j];
            }
            float pos[3], vel[3], so3[3], om[3];
            #pragma unroll
            for (int j = 0; j < 3; ++j) {
                pos[j] = s_xn[r * SDIM + j];     vel[j] = s_xn[r * SDIM + 3 + j];
                so3[j] = s_xn[r * SDIM + 6 + j]; om[j]  = s_xn[r * SDIM + 9 + j];
            }
            float an  = sqrtf(so3[0]*so3[0] + so3[1]*so3[1] + so3[2]*so3[2] + EPS_C);
            float shh = sinf(0.5f * an) / an;
            float qw = cosf(0.5f * an), qx = shh*so3[0], qy = shh*so3[1], qz = shh*so3[2];
            float o2[4], sum2 = 0.0f;
            #pragma unroll
            for (int j = 0; j < 4; ++j) { o2[j] = ue[j]*ue[j]; sum2 += o2[j]; }
            float Tt = KT_C * sum2;
            float tx = KT_C*ARM_C*(o2[2]+o2[3]-o2[0]-o2[1]);
            float ty = KT_C*ARM_C*(o2[1]+o2[2]-o2[0]-o2[3]);
            float tz = KC_C*(o2[0]+o2[2]-o2[1]-o2[3]);
            float ax = 2.0f*(qx*qz + qw*qy) * Tt;
            float ay = 2.0f*(qy*qz - qw*qx) * Tt;
            float az = (1.0f - 2.0f*(qx*qx + qy*qy)) * Tt - GRAV_C;
            const float J0 = 0.01f, J1 = 0.01f, J2 = 0.02f;
            float c0 = om[1]*(om[2]*J2) - om[2]*(om[1]*J1);
            float c1 = om[2]*(om[0]*J0) - om[0]*(om[2]*J2);
            float c2 = om[0]*(om[1]*J1) - om[1]*(om[0]*J0);
            float on0 = om[0] + (tx - c0) / J0 * DT_C;
            float on1 = om[1] + (ty - c1) / J1 * DT_C;
            float on2 = om[2] + (tz - c2) / J2 * DT_C;
            float wdt0 = om[0]*DT_C, wdt1 = om[1]*DT_C, wdt2 = om[2]*DT_C;
            float wn  = sqrtf(wdt0*wdt0 + wdt1*wdt1 + wdt2*wdt2 + EPS_C);
            float sw2 = sinf(0.5f * wn) / wn;
            float dqw = cosf(0.5f * wn), dqx = sw2*wdt0, dqy = sw2*wdt1, dqz = sw2*wdt2;
            float nw_ = qw*dqw - qx*dqx - qy*dqy - qz*dqz;
            float nx_ = qw*dqx + qx*dqw + qy*dqz - qz*dqy;
            float ny_ = qw*dqy - qx*dqz + qy*dqw + qz*dqx;
            float nz_ = qw*dqz + qx*dqy - qy*dqx + qz*dqw;
            float inv = 1.0f / sqrtf(nw_*nw_ + nx_*nx_ + ny_*ny_ + nz_*nz_ + EPS_C);
            nw_ *= inv; nx_ *= inv; ny_ *= inv; nz_ *= inv;
            float inv2 = 1.0f / sqrtf(nw_*nw_ + nx_*nx_ + ny_*ny_ + nz_*nz_ + EPS_C);
            float ww = nw_*inv2, vx = nx_*inv2, vy = ny_*inv2, vz = nz_*inv2;
            float nv = sqrtf(vx*vx + vy*vy + vz*vz + EPS_C);
            float s3 = 2.0f * atan2f(nv, ww) / nv;
            float xp[SDIM] = { pos[0]+vel[0]*DT_C, pos[1]+vel[1]*DT_C, pos[2]+vel[2]*DT_C,
                               vel[0]+ax*DT_C,     vel[1]+ay*DT_C,     vel[2]+az*DT_C,
                               s3*vx, s3*vy, s3*vz, on0, on1, on2 };
            #pragma unroll
            for (int j = 0; j < SDIM; ++j) s_xp[r * SDIM + j] = xp[j];
            for (int j = 0; j < SDIM; ++j)
                s_featA[apack_off(r, j, 5)] = (_Float16)s_xn[r * SDIM + j];
            #pragma unroll
            for (int j = 0; j < CDIM; ++j) {
                s_featA[apack_off(r, 12 + j, 5)] = (_Float16)ur[j];
                s_featA[apack_off(r, 16 + j, 5)] = (_Float16)ue[j];
                s_featA[apack_off(r, 20 + j, 5)] = (_Float16)(ur[j] - ue[j]);
            }
        }
        __syncthreads();

        // MLP1: hid = tanh(feat @ W1 + b1)   (gi/gh now dead -> hidA may alias)
        for (int job = w; job < 128; job += NW) {
            int mt = job >> 6, nt = job & 63;
            const _Float16* ap = s_featA + (size_t)mt * (5 * 512) + lb;
            const _Float16* bp = W1P + ((size_t)nt << 9) + lb;
            v8f acc = {};
            #pragma unroll
            for (int kt = 0; kt < 5; ++kt) {
                v16h a = *(const v16h*)ap;
                v16h b = *(const v16h*)bp;
                acc = wmma_f16(a, b, acc);
                ap += 512;
                bp += 64 * 512;
            }
            int rbase = mt * 16 + ((lane >> 4) << 3);
            int col   = nt * 16 + (lane & 15);
            #pragma unroll
            for (int i = 0; i < 8; ++i)
                s_hidA[apack_off(rbase + i, col, 32)] = (_Float16)tanhf(acc[i] + mlp_b1[col]);
        }
        __syncthreads();

        // MLP2: hid = tanh(hid @ W2 + b2) — 16 live accumulator tiles per wave
        {
            v8f zero = {};
            v8f acc2[2][8];
            #pragma unroll
            for (int m = 0; m < 2; ++m)
                #pragma unroll
                for (int j = 0; j < 8; ++j) acc2[m][j] = zero;

            const _Float16* h0p = s_hidA + lb;               // mt=0, kt=0
            const _Float16* h1p = h0p + 32 * 512;            // mt=1
            const _Float16* w2p = W2P + ((size_t)w << 9) + lb;  // nt = w (+ j*8 via imm)
            #pragma unroll 1
            for (int kt = 0; kt < 32; ++kt) {
                v16h a0 = *(const v16h*)h0p;
                v16h a1 = *(const v16h*)h1p;
                __builtin_prefetch(w2p + 32768, 0, 3);       // next K-slice into L2
                #pragma unroll
                for (int j = 0; j < 8; ++j) {
                    v16h b = *(const v16h*)(w2p + (size_t)j * 4096);  // 8KB imm steps
                    acc2[0][j] = wmma_f16(a0, b, acc2[0][j]);
                    acc2[1][j] = wmma_f16(a1, b, acc2[1][j]);
                }
                h0p += 512;
                h1p += 512;
                w2p += 32768;                                 // 64 tiles * 512
            }
            __syncthreads();   // all waves finished READING hidA
            #pragma unroll
            for (int m = 0; m < 2; ++m)
                #pragma unroll
                for (int j = 0; j < 8; ++j) {
                    int nt = w + j * 8;
                    int rbase = m * 16 + ((lane >> 4) << 3);
                    int col   = nt * 16 + (lane & 15);
                    #pragma unroll
                    for (int i = 0; i < 8; ++i)
                        s_hidA[apack_off(rbase + i, col, 32)] =
                            (_Float16)tanhf(acc2[m][j][i] + mlp_b2[col]);
                }
        }
        __syncthreads();

        // OUT: x_next_cand = x_phys + hid @ out_W + out_b  (waves 0,1; scalar branch)
        if (w < 2) {
            const _Float16* ap = s_hidA + (size_t)w * (32 * 512) + lb;
            const _Float16* bp = WoP + lb;
            v8f acc = {};
            #pragma unroll 1
            for (int kt = 0; kt < 32; ++kt) {
                v16h a = *(const v16h*)ap;
                v16h b = *(const v16h*)bp;
                acc = wmma_f16(a, b, acc);
                ap += 512;
                bp += 512;
            }
            int col = lane & 15;
            if (col < SDIM) {
                int rbase = w * 16 + ((lane >> 4) << 3);
                #pragma unroll
                for (int i = 0; i < 8; ++i)
                    s_xnext[(rbase + i) * SDIM + col] =
                        acc[i] + out_b[col] + s_xp[(rbase + i) * SDIM + col];
            }
        }
        __syncthreads();

        // finalize: finite check per row, commit carry + global output
        if (tid < BT) {
            int r = tid;
            bool fin = true;
            float v[SDIM];
            #pragma unroll
            for (int c = 0; c < SDIM; ++c) {
                v[c] = s_xnext[r * SDIM + c];
                fin = fin && __builtin_isfinite(v[c]);
            }
            #pragma unroll
            for (int c = 0; c < SDIM; ++c) {
                float vv = fin ? v[c] : s_xp[r * SDIM + c];
                s_xn[r * SDIM + c] = vv;
                out[((b0 + r) * T_STEPS + t) * SDIM + c] = vv;
            }
        }
        __syncthreads();
    }
}

// ------------------------------- launcher ----------------------------------
extern "C" void kernel_launch(void* const* d_in, const int* in_sizes, int n_in,
                              void* d_out, int out_size, void* d_ws, size_t ws_size,
                              hipStream_t stream) {
    (void)in_sizes; (void)n_in; (void)out_size; (void)ws_size;
    const float* x0          = (const float*)d_in[0];
    const float* u_seq       = (const float*)d_in[1];
    const float* logit_alpha = (const float*)d_in[2];
    const float* hinit_W     = (const float*)d_in[3];
    const float* hinit_b     = (const float*)d_in[4];
    const float* gru_Wih     = (const float*)d_in[5];
    const float* gru_Whh     = (const float*)d_in[6];
    const float* gru_bih     = (const float*)d_in[7];
    const float* gru_bhh     = (const float*)d_in[8];
    const float* alpha_W     = (const float*)d_in[9];
    const float* alpha_b     = (const float*)d_in[10];
    const float* mlp_W1      = (const float*)d_in[11];
    const float* mlp_b1      = (const float*)d_in[12];
    const float* mlp_W2      = (const float*)d_in[13];
    const float* mlp_b2      = (const float*)d_in[14];
    const float* out_W       = (const float*)d_in[15];
    const float* out_b       = (const float*)d_in[16];

    _Float16* wp = (_Float16*)d_ws;

    auto launch_pack = [&](const float* src, int off, int K, int N, int NT, int total) {
        lagphys_pack_w<<<(total + 255) / 256, 256, 0, stream>>>(src, wp + off, K, N, NT, total);
    };
    launch_pack(gru_Wih, OFF_WIH, 16,   384,  24, 24 * 512);
    launch_pack(gru_Whh, OFF_WHH, 128,  384,  24, 96 * 512);
    launch_pack(mlp_W1,  OFF_W1,  152,  1024, 64, 320 * 512);
    launch_pack(mlp_W2,  OFF_W2,  1024, 1024, 64, 2048 * 512);
    launch_pack(out_W,   OFF_WO,  1024, 12,   1,  32 * 512);

    lagphys_fused_seq<<<B_TOT / BT, 256, 0, stream>>>(
        x0, u_seq, logit_alpha, hinit_W, hinit_b,
        gru_bih, gru_bhh, alpha_W, alpha_b,
        mlp_b1, mlp_b2, out_b, wp, (float*)d_out);
}
